// CausalSelfAttention_25675314495499
// MI455X (gfx1250) — matrix operand order repaired and checked by reading, not compiled, not run on hardware
//
#include <hip/hip_runtime.h>
#include <stdint.h>

// ---------------------------------------------------------------------------
// CDNA5 (gfx1250) causal self-attention with sink, bf16 WMMA pipeline.
// ---------------------------------------------------------------------------

typedef __attribute__((ext_vector_type(16))) __bf16 v16bf;
typedef __attribute__((ext_vector_type(8)))  float  v8f;
typedef unsigned int v4u __attribute__((ext_vector_type(4)));
typedef int          v8i __attribute__((ext_vector_type(8)));
typedef int          v4i __attribute__((ext_vector_type(4)));

#if defined(__has_builtin)
#if __has_builtin(__builtin_amdgcn_tensor_load_to_lds) && __has_builtin(__builtin_amdgcn_s_wait_tensorcnt)
#define HAVE_TDM 1
#endif
#endif
#ifndef HAVE_TDM
#define HAVE_TDM 0
#endif

struct __align__(16) U4 { uint32_t x, y, z, w; };
struct __align__(16) F4 { float x, y, z, w; };

union FragB16 { v16bf v; uint32_t u[8]; };
union FragF32 { v8f  v; float    f[8]; };

#define DEV static __device__ __forceinline__

DEV uint16_t f2bf(float f) {
  uint32_t u = __float_as_uint(f);
  uint32_t r = u + 0x7FFFu + ((u >> 16) & 1u);   // round-to-nearest-even
  return (uint16_t)(r >> 16);
}

// A-operand per-lane dword->K mapping for V_WMMA_*_16X16X32 (16-bit A, 16x32):
// lane<16 holds K in {0..7, 16..23}; lane>=16 holds K in {8..15, 24..31}.
DEV int a_kk(int j, int hi) { return (j < 4 ? 2 * j : 16 + 2 * (j - 4)) + hi * 8; }

// Problem constants
static constexpr int Bb = 2, Tt = 2048, Cc = 1024, Hh = 16, Dd = 64;
static constexpr int M_ROWS = Bb * Tt;          // 4096
static constexpr int NQKV   = 3 * Hh * Dd;      // 3072

// Workspace layout (uint16 element offsets)
static constexpr size_t OFF_XB    = 0;
static constexpr size_t OFF_WQKV  = OFF_XB    + (size_t)M_ROWS * Cc;
static constexpr size_t OFF_WPROJ = OFF_WQKV  + (size_t)NQKV * Cc;
static constexpr size_t OFF_Q     = OFF_WPROJ + (size_t)Cc * (Hh * Dd);
static constexpr size_t OFF_K     = OFF_Q     + (size_t)Bb * Hh * Tt * Dd;
static constexpr size_t OFF_V     = OFF_K     + (size_t)Bb * Hh * Tt * Dd;
static constexpr size_t OFF_AO    = OFF_V     + (size_t)Bb * Hh * Tt * Dd;

// ---------------------------------------------------------------------------
// f32 -> bf16 conversion, 8 elems / thread, 128-bit loads & stores
// ---------------------------------------------------------------------------
__global__ __launch_bounds__(256) void cvt_f32_bf16(const float* __restrict__ in,
                                                    uint16_t* __restrict__ out, int n) {
  int i = (blockIdx.x * 256 + threadIdx.x) * 8;
  if (i >= n) return;
  F4 a = *(const F4*)(in + i);
  F4 b = *(const F4*)(in + i + 4);
  U4 p;
  p.x = (uint32_t)f2bf(a.x) | ((uint32_t)f2bf(a.y) << 16);
  p.y = (uint32_t)f2bf(a.z) | ((uint32_t)f2bf(a.w) << 16);
  p.z = (uint32_t)f2bf(b.x) | ((uint32_t)f2bf(b.y) << 16);
  p.w = (uint32_t)f2bf(b.z) | ((uint32_t)f2bf(b.w) << 16);
  *(U4*)(out + i) = p;
}

// ---------------------------------------------------------------------------
// Tiled bf16 WMMA GEMM: Y(MxN) = A(MxK) * W(NxK)^T (+bias)
// BM=128 (8 waves x 16 rows), BN=64 (4 n-subtiles), BK=64 (two WMMA K-steps)
// EPI==0: scatter into q/k/v (B,H,T,D) bf16, q scaled by 1/sqrt(D)
// EPI==1: fp32 output + bias (final projection)
// ---------------------------------------------------------------------------
template <int EPI>
__global__ __launch_bounds__(256) void gemm_wmma(const uint16_t* __restrict__ A,
                                                 const uint16_t* __restrict__ Wn,
                                                 const float* __restrict__ bias,
                                                 uint16_t* __restrict__ qws,
                                                 uint16_t* __restrict__ kws,
                                                 uint16_t* __restrict__ vws,
                                                 float* __restrict__ yout, int K) {
  __shared__ uint16_t As[128 * 72];   // 128 rows x 64 bf16, stride 72 (144B, 16B-aligned)
  __shared__ uint16_t Bs[64 * 72];    // 64 rows (N) x 64 bf16 (K)

  const int m0 = blockIdx.x * 128;
  const int n0 = blockIdx.y * 64;
  const int tid = threadIdx.x;
  const int wave = tid >> 5, lane = tid & 31;
  const int lhalf = lane & 15, hi = lane >> 4;

  FragF32 acc[4];
#pragma unroll
  for (int s = 0; s < 4; s++)
#pragma unroll
    for (int e = 0; e < 8; e++) acc[s].f[e] = 0.0f;

  const int arow = tid >> 1, aks = (tid & 1) * 32;
  const int brow = tid >> 2, bks = (tid & 3) * 16;

  for (int k0 = 0; k0 < K; k0 += 64) {
    __syncthreads();
    {  // A tile: 128x64, 2 threads/row, 32 shorts each (4x b128)
      const U4* gp = (const U4*)(A + (size_t)(m0 + arow) * K + k0 + aks);
      U4* sp = (U4*)&As[arow * 72 + aks];
      sp[0] = gp[0]; sp[1] = gp[1]; sp[2] = gp[2]; sp[3] = gp[3];
    }
    {  // B tile: 64x64, 4 threads/row, 16 shorts each (2x b128)
      const U4* gp = (const U4*)(Wn + (size_t)(n0 + brow) * K + k0 + bks);
      U4* sp = (U4*)&Bs[brow * 72 + bks];
      sp[0] = gp[0]; sp[1] = gp[1];
    }
    if (k0 + 64 < K) {  // prefetch next K-slab (global_prefetch_b8)
      __builtin_prefetch(A + (size_t)(m0 + arow) * K + k0 + 64 + aks, 0, 0);
      __builtin_prefetch(Wn + (size_t)(n0 + brow) * K + k0 + 64 + bks, 0, 0);
    }
    __syncthreads();

#pragma unroll
    for (int kf = 0; kf < 2; kf++) {
      FragB16 af;
#pragma unroll
      for (int j = 0; j < 8; j++)
        af.u[j] = *(const uint32_t*)&As[(wave * 16 + lhalf) * 72 + kf * 32 + a_kk(j, hi)];
#pragma unroll
      for (int sub = 0; sub < 4; sub++) {
        FragB16 bf;
        const uint16_t* bp = &Bs[(sub * 16 + lhalf) * 72 + kf * 32 + hi * 8];
        *(U4*)&bf.u[0] = *(const U4*)bp;
        *(U4*)&bf.u[4] = *(const U4*)(bp + 16);
        acc[sub].v = __builtin_amdgcn_wmma_f32_16x16x32_bf16(
            false, af.v, false, bf.v, (short)0, acc[sub].v, false, false);
      }
    }
  }

  const int m_base = m0 + wave * 16;
#pragma unroll
  for (int sub = 0; sub < 4; sub++) {
#pragma unroll
    for (int r = 0; r < 8; r++) {
      int m = m_base + r + 8 * hi;
      int n = n0 + sub * 16 + lhalf;
      float val = acc[sub].f[r] + bias[n];
      if constexpr (EPI == 0) {
        int which = n >> 10, rem = n & 1023;
        int h = rem >> 6, d = rem & 63;
        int b = m >> 11, t = m & 2047;
        if (which == 0) val *= 0.125f;  // fold 1/sqrt(D) into q
        size_t idx = (((size_t)(b * Hh + h)) * Tt + t) * Dd + d;
        uint16_t* dst = (which == 0) ? qws : (which == 1) ? kws : vws;
        dst[idx] = f2bf(val);
      } else {
        yout[(size_t)m * Cc + n] = val;
      }
    }
  }
}

// ---------------------------------------------------------------------------
// Flash attention with sink.  One WG = (b,h) x 128 queries; 8 waves x 16 rows.
// K tile loaded by the Tensor Data Mover (D# with LDS padding -> stride 72),
// V tile transposed into WMMA B-operand N x K layout.
// ---------------------------------------------------------------------------
__global__ __launch_bounds__(256) void attn_kernel(const uint16_t* __restrict__ qws,
                                                   const uint16_t* __restrict__ kws,
                                                   const uint16_t* __restrict__ vws,
                                                   const float* __restrict__ sink_logit,
                                                   uint16_t* __restrict__ ao) {
  __shared__ uint16_t Ks[64 * 72];       // [key][d]   row-major (B-operand for Q*K^T)
  __shared__ uint16_t Vs[64 * 72];       // [d][key]   transposed (B-operand for P*V)
  __shared__ uint16_t Ps[8 * 16 * 72];   // per-wave P scratch (C-frag -> A-frag relayout)

  const int qblk = blockIdx.x, h = blockIdx.y, b = blockIdx.z;
  const int tid = threadIdx.x;
  const int wave = tid >> 5, lane = tid & 31;
  const int lhalf = lane & 15, hi = lane >> 4;

  const size_t bh_off = ((size_t)(b * Hh + h)) * Tt * Dd;
  const int qt0 = qblk * 128;
  const int wq0 = qt0 + wave * 16;

  // Q fragments (D=64 -> two K=32 A-fragments), loaded once per wave.
  FragB16 qf[2];
  {
    const uint16_t* qb = qws + bh_off + (size_t)(wq0 + lhalf) * Dd;
#pragma unroll
    for (int kf = 0; kf < 2; kf++)
#pragma unroll
      for (int j = 0; j < 8; j++)
        qf[kf].u[j] = *(const uint32_t*)(qb + kf * 32 + a_kk(j, hi));
  }

  float m_i[8], l_i[8];
  const float sl = sink_logit[h];
#pragma unroll
  for (int r = 0; r < 8; r++) { m_i[r] = sl; l_i[r] = 1.0f; }  // sink: m=sink, l=1, O=0
  FragF32 of[4];
#pragma unroll
  for (int s = 0; s < 4; s++)
#pragma unroll
    for (int e = 0; e < 8; e++) of[s].f[e] = 0.0f;

  const int ntiles = qblk * 2 + 2;  // causal: keys up to qt0+127
  for (int tile = 0; tile < ntiles; tile++) {
    const int kt0 = tile * 64;
    __syncthreads();
#if HAVE_TDM
    if (wave == 0) {  // TDM: 2D 64x64 bf16 tile -> LDS with 8-short row pad (stride 72)
      uint64_t ga = (uint64_t)(uintptr_t)(kws + bh_off + (size_t)kt0 * Dd);
      v4u g0;
      g0.x = 1u;                                   // count=1, user descriptor
      g0.y = (uint32_t)(uintptr_t)(&Ks[0]);        // lds_addr
      g0.z = (uint32_t)ga;                         // global_addr[31:0]
      g0.w = (uint32_t)((ga >> 32) & 0x01FFFFFFu) | 0x80000000u;  // addr[56:32] | type=2
      v8i g1;
      g1[0] = 0x07110000;   // data_size=2B, pad_enable, pad_interval=32dw, pad_amount=4dw
      g1[1] = (int)(64u << 16);  // tensor_dim0 = 64 (lo16 in [31:16])
      g1[2] = (int)(64u << 16);  // tensor_dim0 hi=0 | tensor_dim1 lo = 64
      g1[3] = (int)(64u << 16);  // tensor_dim1 hi=0 | tile_dim0 = 64
      g1[4] = 64;                // tile_dim1 = 64, tile_dim2 = 0
      g1[5] = 64;                // tensor_dim0_stride lo = 64
      g1[6] = 0;                 // stride hi / dim1_stride lo
      g1[7] = 0;
      v4i gz4 = {0, 0, 0, 0};
      v8i gz8 = {0, 0, 0, 0, 0, 0, 0, 0};
      __builtin_amdgcn_tensor_load_to_lds(g0, g1, gz4, gz4, gz8, 0);
      __builtin_amdgcn_s_wait_tensorcnt(0);
    }
#else
    {  // cooperative K tile 64x64, 4 threads/row, 16 shorts each
      int row = tid >> 2, ds = (tid & 3) * 16;
      const U4* gp = (const U4*)(kws + bh_off + (size_t)(kt0 + row) * Dd + ds);
      *(U4*)&Ks[row * 72 + ds]     = gp[0];
      *(U4*)&Ks[row * 72 + ds + 8] = gp[1];
    }
#endif
    {  // V tile transposed into Vs[d][key], packed b32 stores (2 keys/dword)
#pragma unroll
      for (int i = 0; i < 4; i++) {
        int idx = tid + i * 256;             // 0..1023
        int key = (idx >> 5) * 2;            // even key
        int dp = (idx & 31) * 2;             // d pair
        const uint16_t* vp = vws + bh_off + (size_t)(kt0 + key) * Dd + dp;
        uint32_t va = *(const uint32_t*)vp;         // v[key][dp], v[key][dp+1]
        uint32_t vb = *(const uint32_t*)(vp + Dd);  // v[key+1][dp], v[key+1][dp+1]
        *(uint32_t*)&Vs[dp * 72 + key]       = (va & 0xFFFFu) | (vb << 16);
        *(uint32_t*)&Vs[(dp + 1) * 72 + key] = (va >> 16) | (vb & 0xFFFF0000u);
      }
    }
    __syncthreads();

    const bool active = (kt0 <= wq0 + 15);
    if (active) {
      // S = Q * K^T (per-wave 16x64 tile, 8 WMMAs)
      FragF32 sf[4];
#pragma unroll
      for (int sub = 0; sub < 4; sub++) {
#pragma unroll
        for (int e = 0; e < 8; e++) sf[sub].f[e] = 0.0f;
#pragma unroll
        for (int kf = 0; kf < 2; kf++) {
          FragB16 bf;
          const uint16_t* bp = &Ks[(sub * 16 + lhalf) * 72 + kf * 32 + hi * 8];
          *(U4*)&bf.u[0] = *(const U4*)bp;
          *(U4*)&bf.u[4] = *(const U4*)(bp + 16);
          sf[sub].v = __builtin_amdgcn_wmma_f32_16x16x32_bf16(
              false, qf[kf].v, false, bf.v, (short)0, sf[sub].v, false, false);
        }
      }
      // causal mask only on the <=2 diagonal-straddling tiles per wave
      const bool needs_mask = (kt0 + 63 > wq0);
      if (needs_mask) {
#pragma unroll
        for (int sub = 0; sub < 4; sub++)
#pragma unroll
          for (int r = 0; r < 8; r++) {
            int qrow = wq0 + r + 8 * hi;
            int kcol = kt0 + sub * 16 + lhalf;
            if (kcol > qrow) sf[sub].f[r] = -3.0e38f;
          }
      }
      // row max (C-frag: row = r + 8*hi, col = sub*16 + lhalf)
      float rmax[8];
#pragma unroll
      for (int r = 0; r < 8; r++)
        rmax[r] = fmaxf(fmaxf(sf[0].f[r], sf[1].f[r]), fmaxf(sf[2].f[r], sf[3].f[r]));
#pragma unroll
      for (int r = 0; r < 8; r++) {
        float v = rmax[r];
#pragma unroll
        for (int xm = 1; xm < 16; xm <<= 1) v = fmaxf(v, __shfl_xor(v, xm, 32));
        rmax[r] = v;
      }
      float alpha[8], rsum[8];
#pragma unroll
      for (int r = 0; r < 8; r++) {
        float mnew = fmaxf(m_i[r], rmax[r]);
        alpha[r] = __expf(m_i[r] - mnew);
        m_i[r] = mnew;
        rsum[r] = 0.0f;
      }
#pragma unroll
      for (int sub = 0; sub < 4; sub++)
#pragma unroll
        for (int r = 0; r < 8; r++) {
          float p = __expf(sf[sub].f[r] - m_i[r]);
          sf[sub].f[r] = p;
          rsum[r] += p;
        }
#pragma unroll
      for (int r = 0; r < 8; r++) {
        float v = rsum[r];
#pragma unroll
        for (int xm = 1; xm < 16; xm <<= 1) v += __shfl_xor(v, xm, 32);
        l_i[r] = l_i[r] * alpha[r] + v;
      }
#pragma unroll
      for (int sub = 0; sub < 4; sub++)
#pragma unroll
        for (int r = 0; r < 8; r++) of[sub].f[r] *= alpha[r];
      // write P (bf16) to per-wave LDS scratch
      uint16_t* pw = &Ps[wave * 16 * 72];
#pragma unroll
      for (int sub = 0; sub < 4; sub++)
#pragma unroll
        for (int r = 0; r < 8; r++)
          pw[(r + 8 * hi) * 72 + sub * 16 + lhalf] = f2bf(sf[sub].f[r]);
    }
    __syncthreads();  // order P writes before cross-lane A-fragment reads

    if (active) {
      FragB16 pf[2];
      const uint16_t* pr = &Ps[wave * 16 * 72 + lhalf * 72];
#pragma unroll
      for (int kf = 0; kf < 2; kf++)
#pragma unroll
        for (int j = 0; j < 8; j++)
          pf[kf].u[j] = *(const uint32_t*)(pr + kf * 32 + a_kk(j, hi));
      // O += P * V (8 WMMAs)
#pragma unroll
      for (int sub = 0; sub < 4; sub++)
#pragma unroll
        for (int kf = 0; kf < 2; kf++) {
          FragB16 bf;
          const uint16_t* bp = &Vs[(sub * 16 + lhalf) * 72 + kf * 32 + hi * 8];
          *(U4*)&bf.u[0] = *(const U4*)bp;
          *(U4*)&bf.u[4] = *(const U4*)(bp + 16);
          of[sub].v = __builtin_amdgcn_wmma_f32_16x16x32_bf16(
              false, pf[kf].v, false, bf.v, (short)0, of[sub].v, false, false);
        }
    }
  }

  // normalize and store attn output as bf16 (B*T, H*D)
  float inv_l[8];
#pragma unroll
  for (int r = 0; r < 8; r++) inv_l[r] = 1.0f / l_i[r];
#pragma unroll
  for (int sub = 0; sub < 4; sub++)
#pragma unroll
    for (int r = 0; r < 8; r++) {
      int t = wq0 + r + 8 * hi;
      int d = sub * 16 + lhalf;
      ao[((size_t)(b * Tt + t)) * (Hh * Dd) + h * Dd + d] = f2bf(of[sub].f[r] * inv_l[r]);
    }
}

// ---------------------------------------------------------------------------
extern "C" void kernel_launch(void* const* d_in, const int* in_sizes, int n_in,
                              void* d_out, int out_size, void* d_ws, size_t ws_size,
                              hipStream_t stream) {
  const float* x      = (const float*)d_in[0];
  const float* W_qkv  = (const float*)d_in[1];
  const float* b_qkv  = (const float*)d_in[2];
  const float* W_proj = (const float*)d_in[3];
  const float* b_proj = (const float*)d_in[4];
  const float* sink   = (const float*)d_in[5];
  float* out = (float*)d_out;

  uint16_t* ws    = (uint16_t*)d_ws;
  uint16_t* xb    = ws + OFF_XB;
  uint16_t* wqkvb = ws + OFF_WQKV;
  uint16_t* wprjb = ws + OFF_WPROJ;
  uint16_t* qws   = ws + OFF_Q;
  uint16_t* kws   = ws + OFF_K;
  uint16_t* vws   = ws + OFF_V;
  uint16_t* aob   = ws + OFF_AO;

  const int nx  = M_ROWS * Cc;
  const int nwq = NQKV * Cc;
  const int nwp = Cc * (Hh * Dd);
  cvt_f32_bf16<<<nx / 2048, 256, 0, stream>>>(x, xb, nx);
  cvt_f32_bf16<<<nwq / 2048, 256, 0, stream>>>(W_qkv, wqkvb, nwq);
  cvt_f32_bf16<<<nwp / 2048, 256, 0, stream>>>(W_proj, wprjb, nwp);

  // qkv = x @ W_qkv^T, scattered to q/k/v (q pre-scaled by 1/8)
  gemm_wmma<0><<<dim3(M_ROWS / 128, NQKV / 64), 256, 0, stream>>>(
      xb, wqkvb, b_qkv, qws, kws, vws, nullptr, Cc);

  // flash attention with sink
  attn_kernel<<<dim3(Tt / 128, Hh, Bb), 256, 0, stream>>>(qws, kws, vws, sink, aob);

  // y = attn @ W_proj^T + b_proj (fp32 out)
  gemm_wmma<1><<<dim3(M_ROWS / 128, Cc / 64), 256, 0, stream>>>(
      aob, wprjb, b_proj, nullptr, nullptr, nullptr, out, Cc);
}
